// MPNN_41918880809193
// MI455X (gfx1250) — compile-verified, matrix operand order checked by reading, not run on hardware
//
#include <hip/hip_runtime.h>

typedef _Float16 h16;
typedef __attribute__((ext_vector_type(16))) _Float16 v16h;
typedef __attribute__((ext_vector_type(8)))  _Float16 v8h;
typedef __attribute__((ext_vector_type(8)))  float    v8f;

// Sizes: B=4, N=64, D=64, FN=32, FE=16, INTER=128, T=3
// ws layout (bytes):
//   [0, 262144)        wagg_pk : 256 ntile * 32 lane * 16 halves (f16)
//   [262144, 303104)   wr1_pk  : 5 kstep * 8 ntile * 32 * 16 halves
//   [303104, 335872)   wr2_pk  : 4 kstep * 8 ntile * 32 * 16 halves
//   [335872, 401408)   hbuf    : 256*64 f32
//   [401408, 466944)   aggbuf  : 256*64 f32

// ---------------------------------------------------------------------------
// Pack weights into WMMA B-fragment order (f32 -> f16).
// Fragment half layout (matches 16-bit A 16x32 ISA table, mirrored for B):
//   base = (lane<16)?0:8 ;  half h<8 : k = ks*32 + base + h
//                           half h>=8: k = ks*32 + 16 + base + (h-8)
// k >= K_real is zero padding.
// ---------------------------------------------------------------------------
__global__ __launch_bounds__(256) void pack_kernel(
    const float* __restrict__ W_agg, const float* __restrict__ W_r1,
    const float* __restrict__ W_r2,
    h16* __restrict__ wagg_pk, h16* __restrict__ wr1_pk, h16* __restrict__ wr2_pk) {
  int fid = blockIdx.x * 256 + threadIdx.x;   // one fragment (nt,lane) per thread
  const float* src; h16* dst; int lane, nidx, K_real, ksbase, ncols;
  if (fid < 8192) {                 // W_agg: 256 n-tiles, single K-step
    lane = fid & 31; int nt = fid >> 5;
    nidx = nt * 16 + (lane & 15);
    dst = wagg_pk + (size_t)fid * 16; src = W_agg;
    K_real = 16; ksbase = 0; ncols = 4096;
  } else if (fid < 9472) {          // W_r1: 5 ksteps x 8 ntiles
    int f = fid - 8192; lane = f & 31; int unit = f >> 5;
    int ks = unit >> 3, nt = unit & 7;
    nidx = nt * 16 + (lane & 15);
    dst = wr1_pk + (size_t)f * 16; src = W_r1;
    K_real = 144; ksbase = ks * 32; ncols = 128;
  } else {                          // W_r2: 4 ksteps x 8 ntiles
    int f = fid - 9472; lane = f & 31; int unit = f >> 5;
    int ks = unit >> 3, nt = unit & 7;
    nidx = nt * 16 + (lane & 15);
    dst = wr2_pk + (size_t)f * 16; src = W_r2;
    K_real = 128; ksbase = ks * 32; ncols = 128;
  }
  int base = (lane < 16) ? 0 : 8;
#pragma unroll
  for (int hh = 0; hh < 16; ++hh) {
    int k = ksbase + ((hh < 8) ? (base + hh) : (16 + base + (hh - 8)));
    float v = (k < K_real) ? src[(size_t)k * ncols + nidx] : 0.0f;
    dst[hh] = (h16)v;
  }
}

// ---------------------------------------------------------------------------
// h = relu(nodes @ W_en + b_en)   [256 x 32] @ [32 x 64]
// ---------------------------------------------------------------------------
__global__ __launch_bounds__(64) void encode_kernel(
    const float* __restrict__ nodes, const float* __restrict__ W_en,
    const float* __restrict__ b_en, float* __restrict__ hbuf) {
  __shared__ float sn[32];
  int bn = blockIdx.x, d = threadIdx.x;
  if (d < 32) sn[d] = nodes[bn * 32 + d];
  __syncthreads();
  float acc = b_en[d];
#pragma unroll
  for (int k = 0; k < 32; ++k) acc += sn[k] * W_en[k * 64 + d];
  hbuf[bn * 64 + d] = acc > 0.0f ? acc : 0.0f;
}

// ---------------------------------------------------------------------------
// Fused message+aggregate: one workgroup per (b,i).
// agg[b,i,r] = sum_{j active} sum_c relu(E_j @ W_agg + b_agg)[r*64+c] * h[b,j,c]
// Compacts active edges (mask != 0) into m-tiles (f16 in LDS), zero-padded.
// Loop nest: rl (ROLLED: hard cap on in-flight tiles, single scalar acc,
// no register arrays) -> m (dynamic sparsity skip) -> 4 independent WMMAs.
// A-fragment rebuild per (rl,m) is one ds_load_b128 + constant-zero upper.
// ---------------------------------------------------------------------------
__global__ __launch_bounds__(256) void msg_kernel(
    const float* __restrict__ edges, const float* __restrict__ hbuf,
    const v16h* __restrict__ wagg_pk, const float* __restrict__ b_agg,
    float* __restrict__ aggbuf) {
  __shared__ float sE[64][16];                 // raw edge rows for this (b,i)
  __shared__ __align__(16) h16 sEcH[64][16];   // compacted edge rows (f16)
  __shared__ float sMHc[64][64];               // compacted h rows, zero-padded
  __shared__ int   cidx[64];
  __shared__ int   snact;

  int bi = blockIdx.x;             // b*64 + i
  int b  = bi >> 6;
  int i  = bi & 63;
  int t  = threadIdx.x;

  const float* ebase = edges + ((size_t)b * 4096 + (size_t)i * 64) * 16;
  const float* hb    = hbuf + (size_t)b * 4096;

  for (int q = t; q < 1024; q += 256) ((float*)sE)[q] = ebase[q];
  __syncthreads();

  if (t == 0) {
    int c = 0;
    for (int j = 0; j < 64; ++j) if (sE[j][0] != 0.0f) cidx[c++] = j;
    snact = c;
    for (int q = c; q < 64; ++q) cidx[q] = -1;
  }
  __syncthreads();
  int nact   = snact;
  int mtiles = (nact + 15) >> 4;

  for (int q = t; q < 1024; q += 256) {
    int row = q >> 4, f = q & 15;
    int j = cidx[row];
    sEcH[row][f] = (j >= 0) ? (h16)sE[j][f] : (h16)0.0f;
  }
  for (int q = t; q < 4096; q += 256) {
    int row = q >> 6, c = q & 63;
    int j = cidx[row];
    sMHc[row][c] = (j >= 0) ? hb[j * 64 + c] : 0.0f;   // mask==1 for active
  }
  __syncthreads();

  int w = t >> 5, lane = t & 31;
  int lm = lane & 15;
  int lhalf = (lane < 16) ? 0 : 8;

#pragma unroll 1
  for (int rl = 0; rl < 8; ++rl) {           // rolled: caps live registers
    int r = w * 8 + rl;                      // output row, exclusive to wave
    float b0 = b_agg[(r * 4 + 0) * 16 + lm];
    float b1 = b_agg[(r * 4 + 1) * 16 + lm];
    float b2 = b_agg[(r * 4 + 2) * 16 + lm];
    float b3 = b_agg[(r * 4 + 3) * 16 + lm];
    float acc = 0.0f;

#pragma unroll 1
    for (int m = 0; m < mtiles; ++m) {       // dynamic: sparsity skip
      // A fragment: one b128 LDS load (real K=0..15) + zero upper half
      v8h alo = *(const v8h*)&sEcH[m * 16 + lm][lhalf];
      v16h af;
#pragma unroll
      for (int hh = 0; hh < 8; ++hh) af[hh] = alo[hh];
#pragma unroll
      for (int hh = 8; hh < 16; ++hh) af[hh] = (_Float16)0.0f;

      // 4 independent WMMA tiles for this output row
      v16h bf0 = wagg_pk[(r * 4 + 0) * 32 + lane];
      v16h bf1 = wagg_pk[(r * 4 + 1) * 32 + lane];
      v16h bf2 = wagg_pk[(r * 4 + 2) * 32 + lane];
      v16h bf3 = wagg_pk[(r * 4 + 3) * 32 + lane];
      v8f d0 = {}, d1 = {}, d2 = {}, d3 = {};
      d0 = __builtin_amdgcn_wmma_f32_16x16x32_f16(
          false, af, false, bf0, (short)0, d0, false, false);
      d1 = __builtin_amdgcn_wmma_f32_16x16x32_f16(
          false, af, false, bf1, (short)0, d1, false, false);
      d2 = __builtin_amdgcn_wmma_f32_16x16x32_f16(
          false, af, false, bf2, (short)0, d2, false, false);
      d3 = __builtin_amdgcn_wmma_f32_16x16x32_f16(
          false, af, false, bf3, (short)0, d3, false, false);

#pragma unroll
      for (int vv = 0; vv < 8; ++vv) {
        int erow = m * 16 + vv + lhalf;
        // four sMHc reads at 64B-spaced offsets -> ds_load_2addr pairs
        float m0 = sMHc[erow][lm];
        float m1 = sMHc[erow][16 + lm];
        float m2 = sMHc[erow][32 + lm];
        float m3 = sMHc[erow][48 + lm];
        float v0 = d0[vv] + b0; v0 = v0 > 0.0f ? v0 : 0.0f;
        float v1 = d1[vv] + b1; v1 = v1 > 0.0f ? v1 : 0.0f;
        float v2 = d2[vv] + b2; v2 = v2 > 0.0f ? v2 : 0.0f;
        float v3 = d3[vv] + b3; v3 = v3 > 0.0f ? v3 : 0.0f;
        acc += v0 * m0 + v1 * m1 + v2 * m2 + v3 * m3;
      }
    }

    for (int off = 16; off > 0; off >>= 1) acc += __shfl_xor(acc, off, 32);
    if (lane == 0) aggbuf[(size_t)bi * 64 + r] = acc;
  }
}

// ---------------------------------------------------------------------------
// Two fused GRU cells: h1 = GRU(h_old, 0); h = GRU(agg, h1). One block/node.
// ---------------------------------------------------------------------------
__global__ __launch_bounds__(64) void gru_kernel(
    const float* __restrict__ aggbuf, const float* __restrict__ W_ih,
    const float* __restrict__ W_hh, const float* __restrict__ b_ih,
    const float* __restrict__ b_hh, float* __restrict__ hbuf) {
  __shared__ float sh[64], sa[64], sh1[64];
  int bn = blockIdx.x, d = threadIdx.x;
  sh[d] = hbuf[bn * 64 + d];
  sa[d] = aggbuf[bn * 64 + d];
  __syncthreads();

  // cell 1: x = h_old, hprev = 0  =>  gh = b_hh
  float ir = b_ih[d], iz = b_ih[64 + d], in = b_ih[128 + d];
  for (int k = 0; k < 64; ++k) {
    float x = sh[k];
    ir += x * W_ih[d * 64 + k];
    iz += x * W_ih[(64 + d) * 64 + k];
    in += x * W_ih[(128 + d) * 64 + k];
  }
  float r = 1.0f / (1.0f + __expf(-(ir + b_hh[d])));
  float z = 1.0f / (1.0f + __expf(-(iz + b_hh[64 + d])));
  float n = tanhf(in + r * b_hh[128 + d]);
  float h1 = (1.0f - z) * n;
  sh1[d] = h1;
  __syncthreads();

  // cell 2: x = agg, hprev = h1
  float ir2 = b_ih[d], iz2 = b_ih[64 + d], in2 = b_ih[128 + d];
  float hr = b_hh[d], hz = b_hh[64 + d], hn = b_hh[128 + d];
  for (int k = 0; k < 64; ++k) {
    float x = sa[k], hp = sh1[k];
    ir2 += x * W_ih[d * 64 + k];
    iz2 += x * W_ih[(64 + d) * 64 + k];
    in2 += x * W_ih[(128 + d) * 64 + k];
    hr  += hp * W_hh[d * 64 + k];
    hz  += hp * W_hh[(64 + d) * 64 + k];
    hn  += hp * W_hh[(128 + d) * 64 + k];
  }
  float r2 = 1.0f / (1.0f + __expf(-(ir2 + hr)));
  float z2 = 1.0f / (1.0f + __expf(-(iz2 + hz)));
  float n2 = tanhf(in2 + r2 * hn);
  hbuf[bn * 64 + d] = (1.0f - z2) * n2 + z2 * h1;
}

// ---------------------------------------------------------------------------
// Readout MLP over node pairs. One block per 64 pairs (fixed i, all j).
// cat[144] = [h_i(64) | edge(16) | h_j(64)], K padded 144->160.
// GEMM1 (K=160) -> relu -> GEMM2 (K=128) -> relu -> matvec W_r3.
// ---------------------------------------------------------------------------
__global__ __launch_bounds__(256) void readout_kernel(
    const float* __restrict__ edges, const float* __restrict__ hbuf,
    const v16h* __restrict__ wr1_pk, const v16h* __restrict__ wr2_pk,
    const float* __restrict__ b_r1, const float* __restrict__ b_r2,
    const float* __restrict__ W_r3, const float* __restrict__ b_r3,
    float* __restrict__ out) {
  __shared__ __align__(16) h16 sCat[64][160];
  __shared__ __align__(16) h16 sA1[64][128];
  __shared__ __align__(16) h16 sA2[64][128];

  int eb = blockIdx.x;             // b*64 + i
  int b = eb >> 6, chunk = eb & 63;
  int t = threadIdx.x;
  const float* hb = hbuf + (size_t)b * 4096;
  const float* hi = hb + chunk * 64;
  size_t E0 = (size_t)b * 4096 + (size_t)chunk * 64;

  for (int q = t; q < 64 * 160; q += 256) {
    int row = q / 160, col = q % 160;
    float v;
    if (col < 64)        v = hi[col];                               // state_i
    else if (col < 80)   v = edges[(E0 + row) * 16 + (col - 64)];   // edge feats
    else if (col < 144)  v = hb[row * 64 + (col - 80)];             // state_j
    else                 v = 0.0f;                                  // K pad
    sCat[row][col] = (h16)v;
  }
  __syncthreads();

  int w = t >> 5, lane = t & 31, lm = lane & 15;
  int lhalf = (lane < 16) ? 0 : 8;

  // GEMM1: [64 x 160] @ [160 x 128]
#pragma unroll
  for (int q = 0; q < 4; ++q) {
    int id = w * 4 + q, m = id >> 3, n = id & 7;
    v8f acc = {};
#pragma unroll
    for (int ks = 0; ks < 5; ++ks) {
      const h16* ap = &sCat[m * 16 + lm][ks * 32 + lhalf];
      v8h a0 = *(const v8h*)ap;
      v8h a1 = *(const v8h*)(ap + 16);
      v16h af = __builtin_shufflevector(a0, a1, 0,1,2,3,4,5,6,7,8,9,10,11,12,13,14,15);
      v16h bf = wr1_pk[(ks * 8 + n) * 32 + lane];
      acc = __builtin_amdgcn_wmma_f32_16x16x32_f16(
          false, af, false, bf, (short)0, acc, false, false);
    }
    int col = n * 16 + lm;
    float bias = b_r1[col];
#pragma unroll
    for (int vv = 0; vv < 8; ++vv) {
      float val = acc[vv] + bias;
      val = val > 0.0f ? val : 0.0f;
      sA1[m * 16 + vv + lhalf][col] = (h16)val;
    }
  }
  __syncthreads();

  // GEMM2: [64 x 128] @ [128 x 128]
#pragma unroll
  for (int q = 0; q < 4; ++q) {
    int id = w * 4 + q, m = id >> 3, n = id & 7;
    v8f acc = {};
#pragma unroll
    for (int ks = 0; ks < 4; ++ks) {
      const h16* ap = &sA1[m * 16 + lm][ks * 32 + lhalf];
      v8h a0 = *(const v8h*)ap;
      v8h a1 = *(const v8h*)(ap + 16);
      v16h af = __builtin_shufflevector(a0, a1, 0,1,2,3,4,5,6,7,8,9,10,11,12,13,14,15);
      v16h bf = wr2_pk[(ks * 8 + n) * 32 + lane];
      acc = __builtin_amdgcn_wmma_f32_16x16x32_f16(
          false, af, false, bf, (short)0, acc, false, false);
    }
    int col = n * 16 + lm;
    float bias = b_r2[col];
#pragma unroll
    for (int vv = 0; vv < 8; ++vv) {
      float val = acc[vv] + bias;
      val = val > 0.0f ? val : 0.0f;
      sA2[m * 16 + vv + lhalf][col] = (h16)val;
    }
  }
  __syncthreads();

  // matvec: out = sA2 @ W_r3 + b_r3   (4 lanes per row)
  int row = t >> 2, part = t & 3;
  float p = 0.0f;
  const h16* arow = sA2[row];
#pragma unroll
  for (int k = 0; k < 32; ++k) p += (float)arow[part * 32 + k] * W_r3[part * 32 + k];
  p += __shfl_xor(p, 1, 32);
  p += __shfl_xor(p, 2, 32);
  if (part == 0) out[E0 + row] = p + b_r3[0];
}

// ---------------------------------------------------------------------------
extern "C" void kernel_launch(void* const* d_in, const int* in_sizes, int n_in,
                              void* d_out, int out_size, void* d_ws, size_t ws_size,
                              hipStream_t stream) {
  (void)in_sizes; (void)n_in; (void)out_size; (void)ws_size;
  const float* edges = (const float*)d_in[0];
  const float* nodes = (const float*)d_in[1];
  // d_in[2] adjacency: unused by reference
  const float* W_en  = (const float*)d_in[3];
  const float* b_en  = (const float*)d_in[4];
  const float* W_agg = (const float*)d_in[5];
  const float* b_agg = (const float*)d_in[6];
  const float* W_ih  = (const float*)d_in[7];
  const float* W_hh  = (const float*)d_in[8];
  const float* b_ih  = (const float*)d_in[9];
  const float* b_hh  = (const float*)d_in[10];
  const float* W_r1  = (const float*)d_in[11];
  const float* b_r1  = (const float*)d_in[12];
  const float* W_r2  = (const float*)d_in[13];
  const float* b_r2  = (const float*)d_in[14];
  const float* W_r3  = (const float*)d_in[15];
  const float* b_r3  = (const float*)d_in[16];
  float* out = (float*)d_out;

  char* ws = (char*)d_ws;
  h16*   wagg_pk = (h16*)(ws);
  h16*   wr1_pk  = (h16*)(ws + 262144);
  h16*   wr2_pk  = (h16*)(ws + 303104);
  float* hbuf    = (float*)(ws + 335872);
  float* aggbuf  = (float*)(ws + 401408);

  pack_kernel<<<41, 256, 0, stream>>>(W_agg, W_r1, W_r2, wagg_pk, wr1_pk, wr2_pk);
  encode_kernel<<<256, 64, 0, stream>>>(nodes, W_en, b_en, hbuf);
  for (int it = 0; it < 3; ++it) {
    msg_kernel<<<256, 256, 0, stream>>>(edges, hbuf, (const v16h*)wagg_pk, b_agg, aggbuf);
    gru_kernel<<<256, 64, 0, stream>>>(aggbuf, W_ih, W_hh, b_ih, b_hh, hbuf);
  }
  readout_kernel<<<256, 256, 0, stream>>>(edges, hbuf, (const v16h*)wr1_pk,
                                          (const v16h*)wr2_pk, b_r1, b_r2,
                                          W_r3, b_r3, out);
}
